// GIUNetSpect_4320737100489
// MI455X (gfx1250) — matrix operand order repaired and checked by reading, not compiled, not run on hardware
//
#include <hip/hip_runtime.h>

#define BNEPS 1e-5f
#define CDIV(a,b) (((a)+(b)-1)/(b))

typedef __attribute__((ext_vector_type(16))) __bf16 v16bf;
typedef __attribute__((ext_vector_type(16))) unsigned short v16us;
typedef __attribute__((ext_vector_type(8)))  unsigned short v8us;
typedef __attribute__((ext_vector_type(8)))  unsigned int   v8u;
typedef __attribute__((ext_vector_type(8)))  float v8f;

__device__ __forceinline__ unsigned short f2bf(float f) {
  unsigned u = __float_as_uint(f);
  u += 0x7FFFu + ((u >> 16) & 1u);           // round-to-nearest-even
  return (unsigned short)(u >> 16);
}

// pack two floats into one dword holding two bf16 (lo = elem 2i, hi = elem 2i+1)
__device__ __forceinline__ unsigned packbf2(float lo, float hi) {
  unsigned ua = __float_as_uint(lo);
  ua += 0x7FFFu + ((ua >> 16) & 1u);
  unsigned ub = __float_as_uint(hi);
  ub += 0x7FFFu + ((ub >> 16) & 1u);
  return (ua >> 16) | (ub & 0xFFFF0000u);
}

__device__ __forceinline__ float4 add4(float4 a, float4 b) {
  return make_float4(a.x + b.x, a.y + b.y, a.z + b.z, a.w + b.w);
}

// ============ one-shot weight prep: fp32 [K x Nc] row-major -> bf16 [Nc x K] (transposed) ============
__global__ void k_wprep(const float* __restrict__ W, unsigned short* __restrict__ Wb, int K, int Nc) {
  int t = blockIdx.x * blockDim.x + threadIdx.x;
  if (t >= K * Nc) return;
  int c = t / K, k = t - c * K;
  Wb[t] = f2bf(W[(size_t)k * Nc + c]);
}

// ============ WMMA GEMM: Out[MxNc] = (A (+A2))[MxK] @ W[KxNc] + bias (optional ReLU) ============
// Weights pre-converted to bf16, transposed [Nc x K]: each lane's B fragment is two contiguous 8-element
// runs 16 k apart -> two b128 loads. 4 waves/block, each wave a 16-row stripe over NT column tiles
// (A fragment reused across tiles). K multiple of 32, Nc == NT*16. OOB rows load-clamped,
// store-guarded (a bogus A row only affects its own D row, never stored).
template <int NT, bool HASA2>
__global__ void k_wmma_gemm(const float* __restrict__ A, const float* __restrict__ A2,
                            const unsigned short* __restrict__ Wb, const float* __restrict__ bias,
                            float* __restrict__ Out, int M, int K, int Nc, int doRelu) {
  const int lane = threadIdx.x & 31;
  const int wave = threadIdx.x >> 5;
  const int r    = lane & 15;
  const int hi   = lane >> 4;
  const int row0 = (blockIdx.x * 4 + wave) * 16;
  const int row  = row0 + r;
  const int rowc = min(row, M - 1);
  const float4* a4  = (const float4*)(A + (size_t)rowc * K);
  const float4* a24 = HASA2 ? (const float4*)(A2 + (size_t)rowc * K) : nullptr;

  v8f acc[NT] = {};
  for (int k0 = 0; k0 < K; k0 += 32) {
    // A 16x32 bf16 fragment: lanes0-15 elems0-7 -> K=k0..k0+7 (hi half: +8); elems8-15 -> +16
    const int base = (k0 + hi * 8) >> 2;       // float4 index
    float4 p0 = a4[base],     p1 = a4[base + 1];
    float4 q0 = a4[base + 4], q1 = a4[base + 5];
    if (HASA2) {
      p0 = add4(p0, a24[base]);     p1 = add4(p1, a24[base + 1]);
      q0 = add4(q0, a24[base + 4]); q1 = add4(q1, a24[base + 5]);
    }
    v8u au = { packbf2(p0.x, p0.y), packbf2(p0.z, p0.w),
               packbf2(p1.x, p1.y), packbf2(p1.z, p1.w),
               packbf2(q0.x, q0.y), packbf2(q0.z, q0.w),
               packbf2(q1.x, q1.y), packbf2(q1.z, q1.w) };
    const v16bf av = __builtin_bit_cast(v16bf, au);

#pragma unroll
    for (int t = 0; t < NT; ++t) {
      // B 32x16 bf16 fragment: elems0-7 -> K=k0+hi*8..+7, elems8-15 -> +16 (same order as A).
      // Transposed bf16 weights make each 8-element half contiguous per lane.
      const int col = t * 16 + r;
      const v8us* wrow = (const v8us*)(Wb + (size_t)col * K + k0 + hi * 8);
      v8us b0 = wrow[0], b1 = wrow[2];
      v16us bu = __builtin_shufflevector(b0, b1, 0, 1, 2, 3, 4, 5, 6, 7,
                                         8, 9, 10, 11, 12, 13, 14, 15);
      const v16bf bv = __builtin_bit_cast(v16bf, bu);
      acc[t] = __builtin_amdgcn_wmma_f32_16x16x32_bf16(false, av, false, bv,
                                                       (short)0, acc[t], false, false);
    }
  }
#pragma unroll
  for (int t = 0; t < NT; ++t) {
    const int col = t * 16 + r;
    const float bb = bias ? bias[col] : 0.0f;
#pragma unroll
    for (int j = 0; j < 8; ++j) {
      int orow = row0 + hi * 8 + j;  // C/D layout: VGPR j holds rows j and j+8
      if (orow < M) {
        float v = acc[t][j] + bb;
        if (doRelu) v = fmaxf(v, 0.0f);
        Out[(size_t)orow * Nc + col] = v;
      }
    }
  }
}

static void launch_gemm(const float* A, const float* A2, const unsigned short* Wb,
                        const float* bias, float* Out, int M, int K, int Nc, int doRelu,
                        hipStream_t st) {
  dim3 g(CDIV(M, 64));
  const int nt = Nc / 16;
  if (A2) {
    if (nt == 4)      k_wmma_gemm<4, true><<<g, 128, 0, st>>>(A, A2, Wb, bias, Out, M, K, Nc, doRelu);
    else if (nt == 2) k_wmma_gemm<2, true><<<g, 128, 0, st>>>(A, A2, Wb, bias, Out, M, K, Nc, doRelu);
    else              k_wmma_gemm<1, true><<<g, 128, 0, st>>>(A, A2, Wb, bias, Out, M, K, Nc, doRelu);
  } else {
    if (nt == 4)      k_wmma_gemm<4, false><<<g, 128, 0, st>>>(A, A2, Wb, bias, Out, M, K, Nc, doRelu);
    else if (nt == 2) k_wmma_gemm<2, false><<<g, 128, 0, st>>>(A, A2, Wb, bias, Out, M, K, Nc, doRelu);
    else              k_wmma_gemm<1, false><<<g, 128, 0, st>>>(A, A2, Wb, bias, Out, M, K, Nc, doRelu);
  }
}

// ============ BatchNorm (training stats) helpers ============
__global__ void k_col_stats(const float* __restrict__ X, float* __restrict__ stats, int M, int Nc) {
  int c = blockIdx.x;                       // one block per column
  float s = 0.f, s2 = 0.f;
  for (int i = threadIdx.x; i < M; i += blockDim.x) {
    float v = X[(size_t)i * Nc + c];
    s += v; s2 += v * v;
  }
  __shared__ float sh[256], sh2[256];
  sh[threadIdx.x] = s; sh2[threadIdx.x] = s2;
  __syncthreads();
  for (int o = blockDim.x >> 1; o > 0; o >>= 1) {
    if ((int)threadIdx.x < o) { sh[threadIdx.x] += sh[threadIdx.x + o]; sh2[threadIdx.x] += sh2[threadIdx.x + o]; }
    __syncthreads();
  }
  if (threadIdx.x == 0) { stats[c] = sh[0]; stats[Nc + c] = sh2[0]; }
}

__global__ void k_bn_relu(const float* __restrict__ X, const float* __restrict__ stats,
                          const float* __restrict__ g, const float* __restrict__ be,
                          float* __restrict__ Y, int M, int Nc) {
  int idx = blockIdx.x * blockDim.x + threadIdx.x;
  if (idx >= M * Nc) return;
  int c = idx % Nc;
  float mean = stats[c] / (float)M;
  float var  = stats[Nc + c] / (float)M - mean * mean;  // biased var
  float v = (X[idx] - mean) * rsqrtf(var + BNEPS) * g[c] + be[c];
  Y[idx] = fmaxf(v, 0.f);
}

// ============ Graph aggregation (segment_sum over edges), float4-vectorized ============
__global__ void k_edge_scatter(const float* __restrict__ H, const int* __restrict__ src,
                               const int* __restrict__ dst, const float* __restrict__ em,
                               float* __restrict__ agg, int E, int F4) {
  long idx = (long)blockIdx.x * blockDim.x + threadIdx.x;
  if (idx >= (long)E * F4) return;
  int e = (int)(idx / F4);
  int c = (int)(idx - (long)e * F4);
  float w = em ? em[e] : 1.0f;
  if (w == 0.0f) return;
  const int F = F4 * 4;
  float4 v = ((const float4*)(H + (size_t)src[e] * F))[c];
  float* a = agg + (size_t)dst[e] * F + (size_t)c * 4;
  atomicAdd(a + 0, v.x * w);
  atomicAdd(a + 1, v.y * w);
  atomicAdd(a + 2, v.z * w);
  atomicAdd(a + 3, v.w * w);
}

// ============ Laplacian / spectral pooling pieces ============
__global__ void k_deg(const int* src, const int* dst, const float* em, float* deg, int E) {
  int e = blockIdx.x * blockDim.x + threadIdx.x; if (e >= E) return;
  float w = em ? em[e] : 1.0f;
  atomicAdd(&deg[dst[e]], w);
  atomicAdd(&deg[src[e]], w);
}

__global__ void k_dinv(const float* deg, float* dnv, int n) {
  int i = blockIdx.x * blockDim.x + threadIdx.x; if (i >= n) return;
  dnv[i] = rsqrtf(fmaxf(deg[i], 1.0f));
}

__device__ __forceinline__ float hash_unif(unsigned x) {
  x ^= x >> 17; x *= 0xed5ad4bbu; x ^= x >> 11; x *= 0xac4c1b51u;
  x ^= x >> 15; x *= 0x31848babu; x ^= x >> 14;
  return (float)x * (1.0f / 4294967296.0f);
}

__global__ void k_vinit(float* V, int n3) {
  int i = blockIdx.x * blockDim.x + threadIdx.x; if (i >= n3) return;
  unsigned s = (unsigned)i * 2654435761u + 0x9E3779B9u;
  float v = 0.f;
#pragma unroll
  for (int j = 0; j < 4; ++j) v += hash_unif(s + (unsigned)j * 747796405u);
  V[i] = v - 2.0f;                                  // Irwin-Hall approx normal
}

__global__ void k_lmv_scatter(const float* __restrict__ V, const float* __restrict__ dnv,
                              const int* __restrict__ src, const int* __restrict__ dst,
                              const float* __restrict__ em, float* __restrict__ AV, int E) {
  int e = blockIdx.x * blockDim.x + threadIdx.x; if (e >= E) return;
  float w = em ? em[e] : 1.0f;
  if (w == 0.0f) return;
  int s = src[e], d = dst[e];
  float ds = dnv[s] * w, dd = dnv[d] * w;
#pragma unroll
  for (int c = 0; c < 3; ++c) {
    atomicAdd(&AV[d * 3 + c], V[s * 3 + c] * ds);
    atomicAdd(&AV[s * 3 + c], V[d * 3 + c] * dd);
  }
}

__global__ void k_lmv_combine(const float* V, const float* dnv, const float* AV, float* Wv, int n) {
  int i = blockIdx.x * blockDim.x + threadIdx.x; if (i >= n) return;
  float di = dnv[i];
#pragma unroll
  for (int c = 0; c < 3; ++c) Wv[i * 3 + c] = V[i * 3 + c] - di * AV[i * 3 + c];
}

__global__ void k_gram(const float* __restrict__ Wv, float* __restrict__ g6, int n) {
  float acc[6] = {0, 0, 0, 0, 0, 0};
  for (int i = blockIdx.x * blockDim.x + threadIdx.x; i < n; i += gridDim.x * blockDim.x) {
    float w0 = Wv[i * 3], w1 = Wv[i * 3 + 1], w2 = Wv[i * 3 + 2];
    acc[0] += w0 * w0; acc[1] += w0 * w1; acc[2] += w0 * w2;
    acc[3] += w1 * w1; acc[4] += w1 * w2; acc[5] += w2 * w2;
  }
  __shared__ float sh[6][64];
#pragma unroll
  for (int j = 0; j < 6; ++j) sh[j][threadIdx.x] = acc[j];
  __syncthreads();
  for (int o = 32; o > 0; o >>= 1) {
    if ((int)threadIdx.x < o)
#pragma unroll
      for (int j = 0; j < 6; ++j) sh[j][threadIdx.x] += sh[j][threadIdx.x + o];
    __syncthreads();
  }
  if (threadIdx.x == 0)
#pragma unroll
    for (int j = 0; j < 6; ++j) atomicAdd(&g6[j], sh[j][0]);
}

__global__ void k_chol_inv(const float* g6, float* rinv) {
  // QR of n x 3 via Cholesky of Gram: G = R^T R, then store R^{-1} (upper)
  float g0 = g6[0], g1 = g6[1], g2 = g6[2], g3 = g6[3], g4 = g6[4], g5 = g6[5];
  float r00 = sqrtf(fmaxf(g0, 1e-20f));
  float r01 = g1 / r00, r02 = g2 / r00;
  float r11 = sqrtf(fmaxf(g3 - r01 * r01, 1e-20f));
  float r12 = (g4 - r01 * r02) / r11;
  float r22 = sqrtf(fmaxf(g5 - r02 * r02 - r12 * r12, 1e-20f));
  float i00 = 1.f / r00, i11 = 1.f / r11, i22 = 1.f / r22;
  rinv[0] = i00;
  rinv[1] = -r01 * i00 * i11;
  rinv[2] = (r01 * r12 - r02 * r11) * i00 * i11 * i22;
  rinv[3] = i11;
  rinv[4] = -r12 * i11 * i22;
  rinv[5] = i22;
}

__global__ void k_apply_r(const float* Wv, const float* rinv, float* V, int n) {
  int i = blockIdx.x * blockDim.x + threadIdx.x; if (i >= n) return;
  float w0 = Wv[i * 3], w1 = Wv[i * 3 + 1], w2 = Wv[i * 3 + 2];
  V[i * 3]     = w0 * rinv[0];
  V[i * 3 + 1] = w0 * rinv[1] + w1 * rinv[3];
  V[i * 3 + 2] = w0 * rinv[2] + w1 * rinv[4] + w2 * rinv[5];
}

__global__ void k_scores(const float* __restrict__ H, int din, const float* __restrict__ La,
                         const float* Wf, const float* bf, const float* Ws, const float* bs,
                         const float* Wc, const float* bc, float* __restrict__ sc, int n) {
  int i = blockIdx.x * blockDim.x + threadIdx.x; if (i >= n) return;
  float fw = bf[0];
  const float* h = H + (size_t)i * din;
  for (int k = 0; k < din; ++k) fw += h[k] * Wf[k];
  float sw = bs[0] + La[i * 3] * Ws[0] + La[i * 3 + 1] * Ws[1] + La[i * 3 + 2] * Ws[2];
  float z = fw * Wc[0] + sw * Wc[1] + bc[0];
  sc[i] = 1.0f / (1.0f + __expf(-z));
}

// ============ histogram-based top-k set selection (order-free: downstream is scatter-only) ============
__global__ void k_hist(const float* sc, int* hist, int n) {
  int i = blockIdx.x * blockDim.x + threadIdx.x; if (i >= n) return;
  int b = (int)(sc[i] * 1024.0f);
  b = min(max(b, 0), 1023);
  atomicAdd(&hist[b], 1);
}

__global__ void k_pick(const int* hist, int* sel, int k) {
  int cum = 0, b = 1023;
  for (; b >= 0; --b) {
    if (cum + hist[b] >= k) break;
    cum += hist[b];
  }
  if (b < 0) b = 0;
  sel[0] = b;        // threshold bin
  sel[1] = cum;      // strictly-above count
  sel[2] = k - cum;  // take this many from threshold bin
  sel[3] = 0;        // counter: above
  sel[4] = 0;        // counter: boundary
}

__global__ void k_assign(const float* sc, int* perm, int* idxk, int* sel, int n) {
  int i = blockIdx.x * blockDim.x + threadIdx.x; if (i >= n) return;
  int b = min(max((int)(sc[i] * 1024.0f), 0), 1023);
  int thr = sel[0];
  int slot = -1;
  if (b > thr) {
    slot = atomicAdd(&sel[3], 1);
  } else if (b == thr) {
    int pos = atomicAdd(&sel[4], 1);
    if (pos < sel[2]) slot = sel[1] + pos;
  }
  if (slot >= 0) { perm[i] = slot; idxk[slot] = i; }
}

__global__ void k_gather_gate(const float* __restrict__ H, const int* __restrict__ idxk,
                              const float* __restrict__ sc, float* __restrict__ Hp, int k, int F4) {
  long t = (long)blockIdx.x * blockDim.x + threadIdx.x;
  if (t >= (long)k * F4) return;
  int r = (int)(t / F4), c = (int)(t - (long)r * F4);
  int i = idxk[r];
  float g = sc[i];
  const int F = F4 * 4;
  float4 v = ((const float4*)(H + (size_t)i * F))[c];
  ((float4*)(Hp + (size_t)r * F))[c] = make_float4(v.x * g, v.y * g, v.z * g, v.w * g);
}

__global__ void k_remap_edges(const int* src, const int* dst, const float* em, const int* perm,
                              int* ns, int* nd, float* nm, int E) {
  int e = blockIdx.x * blockDim.x + threadIdx.x; if (e >= E) return;
  int a = perm[src[e]], b = perm[dst[e]];
  float w = em ? em[e] : 1.0f;
  nm[e] = (a >= 0 && b >= 0) ? w : 0.0f;
  ns[e] = max(a, 0);
  nd[e] = max(b, 0);
}

__global__ void k_unpool(const float* __restrict__ Hin, const int* __restrict__ idxk,
                         float* __restrict__ Hout, int k, int F4) {
  long t = (long)blockIdx.x * blockDim.x + threadIdx.x;
  if (t >= (long)k * F4) return;
  int r = (int)(t / F4), c = (int)(t - (long)r * F4);
  const int F = F4 * 4;
  ((float4*)(Hout + (size_t)idxk[r] * F))[c] = ((const float4*)(Hin + (size_t)r * F))[c];
}

// ============ final batch mean pool ============
__global__ void k_seg_acc(const float* __restrict__ X, const int* __restrict__ batch,
                          float* __restrict__ out, float* __restrict__ cnt, int n, int C) {
  int i = blockIdx.x * blockDim.x + threadIdx.x; if (i >= n) return;
  int b = batch[i];
  atomicAdd(&cnt[b], 1.0f);
  for (int c = 0; c < C; ++c) atomicAdd(&out[b * C + c], X[(size_t)i * C + c]);
}

__global__ void k_seg_fin(float* out, const float* cnt, int B, int C) {
  int t = threadIdx.x; if (t >= B * C) return;
  out[t] /= fmaxf(cnt[t / C], 1.0f);
}

// ================= host orchestration =================
static void run_gin(const float* Hin, const int* src, const int* dst, const float* em,
                    int n, int din, int dout, int E,
                    const unsigned short* Wb1, const float* b1, const float* g1, const float* be1,
                    const unsigned short* Wb2, const float* b2, const float* g2, const float* be2,
                    float* agg, float* pre, float* tmp, float* stats, float* Hout,
                    hipStream_t st) {
  hipMemsetAsync(agg, 0, (size_t)n * din * 4, st);
  long tot = (long)E * (din / 4);
  k_edge_scatter<<<dim3((unsigned)CDIV(tot, 256)), 256, 0, st>>>(Hin, src, dst, em, agg, E, din / 4);
  launch_gemm(Hin, agg, Wb1, b1, pre, n, din, dout, 0, st);
  k_col_stats<<<dout, 256, 0, st>>>(pre, stats, n, dout);
  k_bn_relu<<<CDIV(n * dout, 256), 256, 0, st>>>(pre, stats, g1, be1, tmp, n, dout);
  launch_gemm(tmp, nullptr, Wb2, b2, pre, n, dout, dout, 0, st);
  k_col_stats<<<dout, 256, 0, st>>>(pre, stats, n, dout);
  k_bn_relu<<<CDIV(n * dout, 256), 256, 0, st>>>(pre, stats, g2, be2, Hout, n, dout);
}

static void run_pool(const float* H, int din, const int* src, const int* dst, const float* em,
                     int n, int k, int E,
                     const float* Wf, const float* bf, const float* Ws, const float* bs,
                     const float* Wc, const float* bc,
                     float* V, float* Wv, float* AV, float* deg, float* dnv, float* sc,
                     float* g6, float* rinv, int* perm, int* idxk, int* hist, int* sel,
                     float* Hp, int* ns, int* nd, float* nm, hipStream_t st) {
  // Laplacian features via orthogonal iteration (QR by Cholesky of 3x3 Gram)
  hipMemsetAsync(deg, 0, (size_t)n * 4, st);
  k_deg<<<CDIV(E, 256), 256, 0, st>>>(src, dst, em, deg, E);
  k_dinv<<<CDIV(n, 256), 256, 0, st>>>(deg, dnv, n);
  k_vinit<<<CDIV(n * 3, 256), 256, 0, st>>>(V, n * 3);
  for (int it = 0; it < 5; ++it) {
    hipMemsetAsync(AV, 0, (size_t)n * 3 * 4, st);
    k_lmv_scatter<<<CDIV(E, 256), 256, 0, st>>>(V, dnv, src, dst, em, AV, E);
    k_lmv_combine<<<CDIV(n, 256), 256, 0, st>>>(V, dnv, AV, Wv, n);
    hipMemsetAsync(g6, 0, 6 * 4, st);
    k_gram<<<240, 64, 0, st>>>(Wv, g6, n);
    k_chol_inv<<<1, 1, 0, st>>>(g6, rinv);
    k_apply_r<<<CDIV(n, 256), 256, 0, st>>>(Wv, rinv, V, n);
  }
  k_scores<<<CDIV(n, 256), 256, 0, st>>>(H, din, V, Wf, bf, Ws, bs, Wc, bc, sc, n);
  // top-k set selection
  hipMemsetAsync(hist, 0, 1024 * 4, st);
  k_hist<<<CDIV(n, 256), 256, 0, st>>>(sc, hist, n);
  k_pick<<<1, 1, 0, st>>>(hist, sel, k);
  hipMemsetAsync(perm, 0xFF, (size_t)n * 4, st);   // perm = -1
  k_assign<<<CDIV(n, 256), 256, 0, st>>>(sc, perm, idxk, sel, n);
  k_gather_gate<<<(unsigned)CDIV((long)k * (din / 4), 256), 256, 0, st>>>(H, idxk, sc, Hp, k, din / 4);
  k_remap_edges<<<CDIV(E, 256), 256, 0, st>>>(src, dst, em, perm, ns, nd, nm, E);
}

extern "C" void kernel_launch(void* const* d_in, const int* in_sizes, int n_in,
                              void* d_out, int out_size, void* d_ws, size_t ws_size,
                              hipStream_t stream) {
  (void)out_size;
  if (n_in < 49) return;
  const float* x = (const float*)d_in[0];
  const float *c1W1 = (const float*)d_in[1],  *c1b1 = (const float*)d_in[2],
              *c1g1 = (const float*)d_in[3],  *c1be1 = (const float*)d_in[4],
              *c1W2 = (const float*)d_in[5],  *c1b2 = (const float*)d_in[6],
              *c1g2 = (const float*)d_in[7],  *c1be2 = (const float*)d_in[8];
  const float *p1Wf = (const float*)d_in[9],  *p1bf = (const float*)d_in[10],
              *p1Ws = (const float*)d_in[11], *p1bs = (const float*)d_in[12],
              *p1Wc = (const float*)d_in[13], *p1bc = (const float*)d_in[14];
  const float *c2W1 = (const float*)d_in[15], *c2b1 = (const float*)d_in[16],
              *c2g1 = (const float*)d_in[17], *c2be1 = (const float*)d_in[18],
              *c2W2 = (const float*)d_in[19], *c2b2 = (const float*)d_in[20],
              *c2g2 = (const float*)d_in[21], *c2be2 = (const float*)d_in[22];
  const float *p2Wf = (const float*)d_in[23], *p2bf = (const float*)d_in[24],
              *p2Ws = (const float*)d_in[25], *p2bs = (const float*)d_in[26],
              *p2Wc = (const float*)d_in[27], *p2bc = (const float*)d_in[28];
  const float *mW1 = (const float*)d_in[29], *mb1 = (const float*)d_in[30],
              *mg1 = (const float*)d_in[31], *mbe1 = (const float*)d_in[32],
              *mW2 = (const float*)d_in[33], *mb2 = (const float*)d_in[34],
              *mg2 = (const float*)d_in[35], *mbe2 = (const float*)d_in[36];
  const float *dW1 = (const float*)d_in[37], *db1 = (const float*)d_in[38],
              *dg1 = (const float*)d_in[39], *dbe1 = (const float*)d_in[40],
              *dW2 = (const float*)d_in[41], *db2 = (const float*)d_in[42],
              *dg2 = (const float*)d_in[43], *dbe2 = (const float*)d_in[44];
  const float* dec1w = (const float*)d_in[45];
  const float* dec1b = (const float*)d_in[46];
  const int* eidx  = (const int*)d_in[47];
  const int* batch = (const int*)d_in[48];

  const int N  = in_sizes[48];
  const int E  = in_sizes[47] / 2;
  const int F  = in_sizes[0] / N;     // 128
  const int K1 = (N * 4) / 5;         // int(0.8*N)
  const int K2 = (K1 * 4) / 5;        // int(0.8*K1)
  const int C  = 16;
  const int B  = 8;
  const int* src = eidx;
  const int* dst = eidx + E;

  // --- workspace carve (256B aligned) ---
  size_t off = 0;
  char* base = (char*)d_ws;
#define ALLOC(T, name, bytes) T* name = (T*)(base + off); off = (off + (size_t)(bytes) + 255) & ~(size_t)255;
  ALLOC(float, agg,  (size_t)N * 128 * 4)
  ALLOC(float, pre,  (size_t)N * 64 * 4)
  ALLOC(float, tmp,  (size_t)N * 64 * 4)
  ALLOC(float, x1,   (size_t)N * 32 * 4)
  ALLOC(float, x1p,  (size_t)K1 * 32 * 4)
  ALLOC(float, x2,   (size_t)K1 * 64 * 4)
  ALLOC(float, x2p,  (size_t)K2 * 64 * 4)
  ALLOC(float, xm,   (size_t)K2 * 64 * 4)
  ALLOC(float, xd2,  (size_t)K1 * 32 * 4)
  ALLOC(float, Vv,   (size_t)N * 3 * 4)
  ALLOC(float, Wv,   (size_t)N * 3 * 4)
  ALLOC(float, AV,   (size_t)N * 3 * 4)
  ALLOC(float, deg,  (size_t)N * 4)
  ALLOC(float, dnv,  (size_t)N * 4)
  ALLOC(float, sc,   (size_t)N * 4)
  ALLOC(float, stats, 128 * 4)
  ALLOC(float, g6,    64 * 4)
  ALLOC(float, rinv,  64 * 4)
  ALLOC(int,   perm, (size_t)N * 4)
  ALLOC(int,   idx1, (size_t)K1 * 4)
  ALLOC(int,   idx2, (size_t)K2 * 4)
  ALLOC(int,   s1,   (size_t)E * 4)
  ALLOC(int,   d1,   (size_t)E * 4)
  ALLOC(float, m1,   (size_t)E * 4)
  ALLOC(int,   s2,   (size_t)E * 4)
  ALLOC(int,   d2,   (size_t)E * 4)
  ALLOC(float, m2,   (size_t)E * 4)
  ALLOC(int,   hist, 1024 * 4)
  ALLOC(int,   sel,  64 * 4)
  ALLOC(float, cnt,  64 * 4)
  ALLOC(unsigned short, wbC1a, (size_t)F * 32 * 2)
  ALLOC(unsigned short, wbC1b, (size_t)32 * 32 * 2)
  ALLOC(unsigned short, wbC2a, (size_t)32 * 64 * 2)
  ALLOC(unsigned short, wbC2b, (size_t)64 * 64 * 2)
  ALLOC(unsigned short, wbMa,  (size_t)64 * 64 * 2)
  ALLOC(unsigned short, wbMb,  (size_t)64 * 64 * 2)
  ALLOC(unsigned short, wbDa,  (size_t)64 * 32 * 2)
  ALLOC(unsigned short, wbDb,  (size_t)32 * 32 * 2)
  ALLOC(unsigned short, wbD1,  (size_t)32 * 16 * 2)
#undef ALLOC
  if (off > ws_size) return;
  float* xd2b = x2;    // reuse: x2 dead after pool2
  float* xd1f = x1;    // reuse: x1 dead after pool1
  float* xd1  = x1p;   // reuse: x1p dead after conv2 (N*16*4 <= K1*32*4)

  // weight prep: fp32 [KxNc] -> bf16 [NcxK]
  k_wprep<<<CDIV(F * 32, 256), 256, 0, stream>>>(c1W1, wbC1a, F, 32);
  k_wprep<<<CDIV(32 * 32, 256), 256, 0, stream>>>(c1W2, wbC1b, 32, 32);
  k_wprep<<<CDIV(32 * 64, 256), 256, 0, stream>>>(c2W1, wbC2a, 32, 64);
  k_wprep<<<CDIV(64 * 64, 256), 256, 0, stream>>>(c2W2, wbC2b, 64, 64);
  k_wprep<<<CDIV(64 * 64, 256), 256, 0, stream>>>(mW1,  wbMa,  64, 64);
  k_wprep<<<CDIV(64 * 64, 256), 256, 0, stream>>>(mW2,  wbMb,  64, 64);
  k_wprep<<<CDIV(64 * 32, 256), 256, 0, stream>>>(dW1,  wbDa,  64, 32);
  k_wprep<<<CDIV(32 * 32, 256), 256, 0, stream>>>(dW2,  wbDb,  32, 32);
  k_wprep<<<CDIV(32 * 16, 256), 256, 0, stream>>>(dec1w, wbD1, 32, 16);

  // encoder
  run_gin(x, src, dst, nullptr, N, F, 32, E,
          wbC1a, c1b1, c1g1, c1be1, wbC1b, c1b2, c1g2, c1be2,
          agg, pre, tmp, stats, x1, stream);
  run_pool(x1, 32, src, dst, nullptr, N, K1, E,
           p1Wf, p1bf, p1Ws, p1bs, p1Wc, p1bc,
           Vv, Wv, AV, deg, dnv, sc, g6, rinv, perm, idx1, hist, sel,
           x1p, s1, d1, m1, stream);
  run_gin(x1p, s1, d1, m1, K1, 32, 64, E,
          wbC2a, c2b1, c2g1, c2be1, wbC2b, c2b2, c2g2, c2be2,
          agg, pre, tmp, stats, x2, stream);
  run_pool(x2, 64, s1, d1, m1, K1, K2, E,
           p2Wf, p2bf, p2Ws, p2bs, p2Wc, p2bc,
           Vv, Wv, AV, deg, dnv, sc, g6, rinv, perm, idx2, hist, sel,
           x2p, s2, d2, m2, stream);
  run_gin(x2p, s2, d2, m2, K2, 64, 64, E,
          wbMa, mb1, mg1, mbe1, wbMb, mb2, mg2, mbe2,
          agg, pre, tmp, stats, xm, stream);

  // decoder
  hipMemsetAsync(xd2b, 0, (size_t)K1 * 64 * 4, stream);
  k_unpool<<<(unsigned)CDIV((long)K2 * 16, 256), 256, 0, stream>>>(xm, idx2, xd2b, K2, 16);
  run_gin(xd2b, s2, d2, m2, K1, 64, 32, E,
          wbDa, db1, dg1, dbe1, wbDb, db2, dg2, dbe2,
          agg, pre, tmp, stats, xd2, stream);
  hipMemsetAsync(xd1f, 0, (size_t)N * 32 * 4, stream);
  k_unpool<<<(unsigned)CDIV((long)K1 * 8, 256), 256, 0, stream>>>(xd2, idx1, xd1f, K1, 8);
  launch_gemm(xd1f, nullptr, wbD1, dec1b, xd1, N, 32, C, 1, stream);

  // batch mean pool
  hipMemsetAsync(d_out, 0, (size_t)B * C * 4, stream);
  hipMemsetAsync(cnt, 0, (size_t)B * 4, stream);
  k_seg_acc<<<CDIV(N, 256), 256, 0, stream>>>(xd1, batch, (float*)d_out, cnt, N, C);
  k_seg_fin<<<1, 128, 0, stream>>>((float*)d_out, cnt, B, C);
}
